// GrabPhisher_22617297781310
// MI455X (gfx1250) — matrix operand order repaired and checked by reading, compiled, not run on hardware
//
#include <hip/hip_runtime.h>
#include <hip/hip_bf16.h>

// ---------------- problem constants (fixed by the reference) ----------------
#define N_      50000
#define E_      400000
#define EE_     450000      // E + N self loops
#define DIN_    32
#define HID_    256
#define HEADS_  4
#define C_      64
#define G_      64
#define OUT_    2

typedef __attribute__((ext_vector_type(16))) _Float16 v16h;
typedef __attribute__((ext_vector_type(8)))  float    v8f;

// ------------- order-preserving float<->uint for atomic segment-max ---------
__device__ __forceinline__ unsigned ford(float f) {
  unsigned u = __float_as_uint(f);
  return (u & 0x80000000u) ? ~u : (u | 0x80000000u);
}
__device__ __forceinline__ float finv(unsigned u) {
  return __uint_as_float((u & 0x80000000u) ? (u & 0x7FFFFFFFu) : ~u);
}

__device__ __forceinline__ void edge_sd(const int* __restrict__ ei, int e, int& s, int& d) {
  if (e < E_) { s = ei[e]; d = ei[E_ + e]; }
  else        { s = d = e - E_; }              // self loops appended
}

// ---------------------------------------------------------------------------
// Fragment packing (ISA 7.12.2 wave32 layouts), f32 -> f16.
// A[M,K] -> Ap tiles (M/16)x(K/32); per tile 32 lanes x 16 f16 (32B/lane).
//   lanes 0-15:  M=lane,    K = {0..7} and {16..23}  (pairs)
//   lanes 16-31: M=lane-16, K = {8..15} and {24..31}
// ---------------------------------------------------------------------------
__global__ void gp_pack_a(const float* __restrict__ A, _Float16* __restrict__ Ap,
                          int M, int K) {
  int tid  = blockIdx.x * blockDim.x + threadIdx.x;      // (tile, lane)
  int tiles = (M >> 4) * (K >> 5);
  if (tid >= tiles * 32) return;
  int lane = tid & 31, tile = tid >> 5;
  int ktiles = K >> 5;
  int mt = tile / ktiles, kt = tile % ktiles;
  int m  = (mt << 4) + (lane & 15);
  int koff = (lane >> 4) << 3;                           // 0 or 8
  const float* arow = A + (size_t)m * K + (kt << 5);
  v16h a;
  #pragma unroll
  for (int v = 0; v < 4; ++v) {
    float2 lo = *(const float2*)(arow + koff + 2 * v);
    float2 hi = *(const float2*)(arow + 16 + koff + 2 * v);
    a[2 * v]     = (_Float16)lo.x;  a[2 * v + 1]     = (_Float16)lo.y;
    a[8 + 2 * v] = (_Float16)hi.x;  a[8 + 2 * v + 1] = (_Float16)hi.y;
  }
  *(v16h*)(Ap + (size_t)tid * 16) = a;                   // 32B contiguous store
}

// B[K,N] -> Bp tiles (K/32)x(N/16); per tile 32 lanes x 16 f16.
//   lanes 0-15: N=lane, K={0..7,16..23}; lanes 16-31: N=lane-16, K={8..15,24..31}
__global__ void gp_pack_b(const float* __restrict__ B, _Float16* __restrict__ Bp,
                          int K, int N) {
  int tid  = blockIdx.x * blockDim.x + threadIdx.x;
  int tiles = (K >> 5) * (N >> 4);
  if (tid >= tiles * 32) return;
  int lane = tid & 31, tile = tid >> 5;
  int ntiles = N >> 4;
  int kt = tile / ntiles, nt = tile % ntiles;
  int n  = (nt << 4) + (lane & 15);
  int kbase = (kt << 5) + ((lane >> 4) << 3);            // +0 or +8
  v16h b;
  #pragma unroll
  for (int v = 0; v < 8; ++v) {
    b[2 * v]     = (_Float16)B[(size_t)(kbase + 2 * v + (v >= 4 ? 8 : 0)) * N + n];
    b[2 * v + 1] = (_Float16)B[(size_t)(kbase + 2 * v + 1 + (v >= 4 ? 8 : 0)) * N + n];
  }
  *(v16h*)(Bp + (size_t)tid * 16) = b;
}

// ---------------------------------------------------------------------------
// WMMA GEMM on packed fragments: C[M,N] = A@B (+bias), f32 accumulate.
// One wave per 16x64 strip; A fragment reused across 4 WMMAs per K-step.
// Per K-step: 1x32B A load + 4x32B B loads + 4 wmma (all b128-class vmem).
// ---------------------------------------------------------------------------
template <int K, int N>
__global__ __launch_bounds__(128)
void gp_gemm_wmma(const _Float16* __restrict__ Ap, const _Float16* __restrict__ Bp,
                  const float* __restrict__ bias, float* __restrict__ C, int M) {
  constexpr int KT = K >> 5;                    // k tiles
  constexpr int NT = N >> 4;                    // n tiles
  const int lane  = threadIdx.x & 31;
  const int wave  = blockIdx.x * (blockDim.x >> 5) + (threadIdx.x >> 5);
  const int ntn   = N >> 6;                     // 64-col strips
  const int total = (M >> 4) * ntn;
  if (wave >= total) return;                    // wave-uniform: EXEC stays all-1
  const int mt = wave / ntn, nw = wave % ntn;
  const int nt0 = nw << 2;

  const _Float16* apw = Ap + (((size_t)mt * KT) * 32 + lane) * 16;
  const _Float16* bpw = Bp + ((size_t)nt0 * 32 + lane) * 16;

  v8f acc[4] = {v8f{}, v8f{}, v8f{}, v8f{}};

  #pragma unroll 4
  for (int kt = 0; kt < KT; ++kt) {
    v16h a = *(const v16h*)(apw + (size_t)kt * 32 * 16);
    if (kt + 1 < KT)                            // hint: global_prefetch_b8
      __builtin_prefetch(apw + (size_t)(kt + 1) * 32 * 16, 0, 1);
    const _Float16* bk = bpw + (size_t)kt * NT * 32 * 16;
    #pragma unroll
    for (int s = 0; s < 4; ++s) {
      v16h b = *(const v16h*)(bk + (size_t)s * 32 * 16);
      acc[s] = __builtin_amdgcn_wmma_f32_16x16x32_f16(
          false, a, false, b, (short)0, acc[s], false, false);
    }
  }

  // C/D layout: VGPR r -> M = r + (lane<16 ? 0 : 8), N = lane%16
  const int mbase = (mt << 4) + ((lane >> 4) << 3);
  const int ncol  = lane & 15;
  #pragma unroll
  for (int s = 0; s < 4; ++s) {
    const int cn = (nw << 6) + (s << 4) + ncol;
    const float bv = bias ? bias[cn] : 0.0f;
    #pragma unroll
    for (int r = 0; r < 8; ++r)
      C[(size_t)(mbase + r) * N + cn] = acc[s][r] + bv;
  }
}

// ---------------- per-node attention scores es/ed (dot over C=64) ----------
__global__ void gp_attn_scores(const float* __restrict__ hw,
                               const float* __restrict__ asrc,
                               const float* __restrict__ adst,
                               float* __restrict__ es, float* __restrict__ ed) {
  int i = blockIdx.x * blockDim.x + threadIdx.x;
  if (i >= N_ * HEADS_) return;
  int n = i >> 2, hh = i & 3;
  const float* base = hw + (size_t)n * HID_ + hh * C_;
  const float* as = asrc + hh * C_;
  const float* ad = adst + hh * C_;
  float s = 0.f, d = 0.f;
  #pragma unroll 4
  for (int c = 0; c < C_; c += 4) {
    float4 hv = *(const float4*)(base + c);
    float4 av = *(const float4*)(as + c);
    float4 dv = *(const float4*)(ad + c);
    s += hv.x * av.x + hv.y * av.y + hv.z * av.z + hv.w * av.w;
    d += hv.x * dv.x + hv.y * dv.y + hv.z * dv.z + hv.w * dv.w;
  }
  es[i] = s; ed[i] = d;
}

// ---------------- edge pass 1: leaky_relu(e) + segment max over dst --------
__global__ void gp_edge_p1(const int* __restrict__ ei,
                           const float* __restrict__ es, const float* __restrict__ ed,
                           float* __restrict__ ebuf, unsigned* __restrict__ emax_u) {
  int e = blockIdx.x * blockDim.x + threadIdx.x;
  if (e >= EE_) return;
  int s, d; edge_sd(ei, e, s, d);
  #pragma unroll
  for (int h = 0; h < HEADS_; ++h) {
    float v = es[s * HEADS_ + h] + ed[d * HEADS_ + h];
    v = v > 0.f ? v : 0.2f * v;                   // leaky_relu(0.2)
    ebuf[(size_t)e * HEADS_ + h] = v;
    atomicMax(&emax_u[d * HEADS_ + h], ford(v));
  }
}

// ---------------- edge pass 2: exp(e - max) + segment sum ------------------
__global__ void gp_edge_p2(const int* __restrict__ ei,
                           float* __restrict__ ebuf,
                           const unsigned* __restrict__ emax_u,
                           float* __restrict__ denom) {
  int e = blockIdx.x * blockDim.x + threadIdx.x;
  if (e >= EE_) return;
  int s, d; edge_sd(ei, e, s, d);
  #pragma unroll
  for (int h = 0; h < HEADS_; ++h) {
    float m  = finv(emax_u[d * HEADS_ + h]);      // every dst has a self loop
    float ex = __expf(ebuf[(size_t)e * HEADS_ + h] - m);
    ebuf[(size_t)e * HEADS_ + h] = ex;
    atomicAdd(&denom[d * HEADS_ + h], ex);
  }
}

// ---------------- edge pass 3: out[dst] += alpha * hw[src] -----------------
// 4 edges per 256-thread block; 64 threads/edge, 4 channels/thread.
__global__ void gp_edge_p3(const int* __restrict__ ei,
                           const float* __restrict__ ebuf,
                           const float* __restrict__ denom,
                           const float* __restrict__ hw,
                           float* __restrict__ outacc) {
  int e = blockIdx.x * 4 + (threadIdx.x >> 6);
  if (e >= EE_) return;
  int t = threadIdx.x & 63;
  int s, d; edge_sd(ei, e, s, d);
  int c0 = t * 4;
  int h  = c0 >> 6;                               // head = channel/64
  float alpha = ebuf[(size_t)e * HEADS_ + h] /
                (denom[d * HEADS_ + h] + 1e-16f);
  float4 hv = *(const float4*)&hw[(size_t)s * HID_ + c0];
  float* o  = &outacc[(size_t)d * HID_ + c0];
  atomicAdd(o + 0, hv.x * alpha);
  atomicAdd(o + 1, hv.y * alpha);
  atomicAdd(o + 2, hv.z * alpha);
  atomicAdd(o + 3, hv.w * alpha);
}

// ---------------- h = elu(outacc + bias) -----------------------------------
__global__ void gp_bias_elu(const float* __restrict__ outacc,
                            const float* __restrict__ bias,
                            float* __restrict__ h) {
  int i = blockIdx.x * blockDim.x + threadIdx.x;
  if (i >= N_ * HID_) return;
  float v = outacc[i] + bias[i & (HID_ - 1)];
  h[i] = v > 0.f ? v : (__expf(v) - 1.0f);
}

// ---------------- graph mean pooling ---------------------------------------
__global__ void gp_count(const int* __restrict__ batch, float* __restrict__ counts) {
  int n = blockIdx.x * blockDim.x + threadIdx.x;
  if (n < N_) atomicAdd(&counts[batch[n]], 1.0f);
}
__global__ void gp_pool_sum(const float* __restrict__ h, const int* __restrict__ batch,
                            float* __restrict__ gemb) {
  int n = blockIdx.x;
  int c = threadIdx.x * 4;                        // 64 threads
  int g = batch[n];
  float4 v = *(const float4*)&h[(size_t)n * HID_ + c];
  float* o = &gemb[g * HID_ + c];
  atomicAdd(o + 0, v.x); atomicAdd(o + 1, v.y);
  atomicAdd(o + 2, v.z); atomicAdd(o + 3, v.w);
}
__global__ void gp_pool_div(float* __restrict__ gemb, const float* __restrict__ counts) {
  int i = blockIdx.x * blockDim.x + threadIdx.x;
  if (i < G_ * HID_) gemb[i] /= fmaxf(counts[i >> 8], 1.0f);
}

// ---------------- GRU layer over the 64-step sequence (single block) -------
__global__ __launch_bounds__(HID_)
void gp_gru(const float* __restrict__ xs, const float* __restrict__ w_ih,
            const float* __restrict__ w_hh, const float* __restrict__ b_ih,
            const float* __restrict__ b_hh, float* __restrict__ ys) {
  __shared__ float xt[HID_], ht[HID_];
  int j = threadIdx.x;
  float h = 0.0f;
  for (int t = 0; t < G_; ++t) {
    ht[j] = h;
    xt[j] = xs[t * HID_ + j];
    __syncthreads();
    float ir = b_ih[j], iz = b_ih[HID_ + j], in_ = b_ih[2 * HID_ + j];
    float hr = b_hh[j], hz = b_hh[HID_ + j], hn  = b_hh[2 * HID_ + j];
    const float* wr = w_ih + (size_t)j * HID_;
    const float* wz = w_ih + (size_t)(HID_ + j) * HID_;
    const float* wn = w_ih + (size_t)(2 * HID_ + j) * HID_;
    const float* ur = w_hh + (size_t)j * HID_;
    const float* uz = w_hh + (size_t)(HID_ + j) * HID_;
    const float* un = w_hh + (size_t)(2 * HID_ + j) * HID_;
    for (int k = 0; k < HID_; ++k) {
      float xv = xt[k], hv = ht[k];
      ir += xv * wr[k]; iz += xv * wz[k]; in_ += xv * wn[k];
      hr += hv * ur[k]; hz += hv * uz[k]; hn  += hv * un[k];
    }
    float r  = 1.0f / (1.0f + __expf(-(ir + hr)));
    float z  = 1.0f / (1.0f + __expf(-(iz + hz)));
    float nn = tanhf(in_ + r * hn);
    h = (1.0f - z) * nn + z * h;
    ys[t * HID_ + j] = h;
    __syncthreads();
  }
}

// ---------------- classifier: relu(seq@cw1+cb1)@cw2+cb2 --------------------
__global__ __launch_bounds__(128)
void gp_classifier(const float* __restrict__ seq, const float* __restrict__ cw1,
                   const float* __restrict__ cb1, const float* __restrict__ cw2,
                   const float* __restrict__ cb2, float* __restrict__ out) {
  __shared__ float hid[HID_ / 2];
  int g = blockIdx.x, t = threadIdx.x;
  float acc = cb1[t];
  for (int k = 0; k < HID_; ++k)
    acc += seq[g * HID_ + k] * cw1[k * (HID_ / 2) + t];
  hid[t] = fmaxf(acc, 0.0f);
  __syncthreads();
  if (t < OUT_) {
    float o = cb2[t];
    for (int k = 0; k < HID_ / 2; ++k) o += hid[k] * cw2[k * OUT_ + t];
    out[g * OUT_ + t] = o;
  }
}

// ===========================================================================
extern "C" void kernel_launch(void* const* d_in, const int* in_sizes, int n_in,
                              void* d_out, int out_size, void* d_ws, size_t ws_size,
                              hipStream_t stream) {
  const float* x        = (const float*)d_in[0];
  const int*   ei       = (const int*)  d_in[1];
  const int*   batch    = (const int*)  d_in[2];
  const float* w_proj   = (const float*)d_in[4];
  const float* b_proj   = (const float*)d_in[5];
  const float* gat_w    = (const float*)d_in[6];
  const float* gat_asrc = (const float*)d_in[7];
  const float* gat_adst = (const float*)d_in[8];
  const float* gat_b    = (const float*)d_in[9];
  const float* gru_wih  = (const float*)d_in[10];
  const float* gru_whh  = (const float*)d_in[11];
  const float* gru_bih  = (const float*)d_in[12];
  const float* gru_bhh  = (const float*)d_in[13];
  const float* cw1      = (const float*)d_in[14];
  const float* cb1      = (const float*)d_in[15];
  const float* cw2      = (const float*)d_in[16];
  const float* cb2      = (const float*)d_in[17];
  float* out = (float*)d_out;

  // ---- workspace layout ----
  char* ws = (char*)d_ws;
  size_t off = 0;
  auto take = [&](size_t bytes) -> char* {
    char* p = ws + off;
    off += (bytes + 255) & ~(size_t)255;
    return p;
  };
  float*     h      = (float*)    take((size_t)N_ * HID_ * 4);
  float*     hw     = (float*)    take((size_t)N_ * HID_ * 4);
  float*     outacc = (float*)    take((size_t)N_ * HID_ * 4);
  _Float16*  Ap     = (_Float16*) take((size_t)N_ * HID_ * 2);   // packed A frags
  _Float16*  Bp     = (_Float16*) take((size_t)HID_ * HID_ * 2); // packed B frags
  float*     es     = (float*)    take((size_t)N_ * HEADS_ * 4);
  float*     ed     = (float*)    take((size_t)N_ * HEADS_ * 4);
  unsigned*  emax_u = (unsigned*) take((size_t)N_ * HEADS_ * 4);
  float*     denom  = (float*)    take((size_t)N_ * HEADS_ * 4);
  float*     ebuf   = (float*)    take((size_t)EE_ * HEADS_ * 4);
  float*     gemb   = (float*)    take((size_t)G_ * HID_ * 4);
  float*     counts = (float*)    take((size_t)G_ * 4);
  float*     seqB   = (float*)    take((size_t)G_ * HID_ * 4);
  float*     seqC   = (float*)    take((size_t)G_ * HID_ * 4);

  const int gemmWaves  = (N_ / 16) * (HID_ / 64);      // 12500
  const int gemmBlocks = (gemmWaves + 3) / 4;          // 128 thr = 4 waves
  const int edgeB256   = (EE_ + 255) / 256;

  // ---- input projection: h = x @ w_proj + b_proj ----
  {
    int at = (N_ / 16) * (DIN_ / 32) * 32;             // pack-a threads
    int bt = (DIN_ / 32) * (HID_ / 16) * 32;           // pack-b threads
    gp_pack_a<<<(at + 255) / 256, 256, 0, stream>>>(x, Ap, N_, DIN_);
    gp_pack_b<<<(bt + 255) / 256, 256, 0, stream>>>(w_proj, Bp, DIN_, HID_);
    gp_gemm_wmma<DIN_, HID_><<<gemmBlocks, 128, 0, stream>>>(Ap, Bp, b_proj, h, N_);
  }

  // ---- 3 GAT layers ----
  const int atH = (N_ / 16) * (HID_ / 32) * 32;
  const int btH = (HID_ / 32) * (HID_ / 16) * 32;
  for (int l = 0; l < 3; ++l) {
    gp_pack_a<<<(atH + 255) / 256, 256, 0, stream>>>(h, Ap, N_, HID_);
    gp_pack_b<<<(btH + 255) / 256, 256, 0, stream>>>(
        gat_w + (size_t)l * HID_ * HID_, Bp, HID_, HID_);
    gp_gemm_wmma<HID_, HID_><<<gemmBlocks, 128, 0, stream>>>(Ap, Bp, nullptr, hw, N_);
    gp_attn_scores<<<(N_ * HEADS_ + 255) / 256, 256, 0, stream>>>(
        hw, gat_asrc + l * HEADS_ * C_, gat_adst + l * HEADS_ * C_, es, ed);
    hipMemsetAsync(emax_u, 0, (size_t)N_ * HEADS_ * 4, stream);
    hipMemsetAsync(denom,  0, (size_t)N_ * HEADS_ * 4, stream);
    hipMemsetAsync(outacc, 0, (size_t)N_ * HID_ * 4, stream);
    gp_edge_p1<<<edgeB256, 256, 0, stream>>>(ei, es, ed, ebuf, emax_u);
    gp_edge_p2<<<edgeB256, 256, 0, stream>>>(ei, ebuf, emax_u, denom);
    gp_edge_p3<<<(EE_ + 3) / 4, 256, 0, stream>>>(ei, ebuf, denom, hw, outacc);
    gp_bias_elu<<<(N_ * HID_ + 255) / 256, 256, 0, stream>>>(
        outacc, gat_b + l * HID_, h);
  }

  // ---- graph mean pooling ----
  hipMemsetAsync(gemb,   0, (size_t)G_ * HID_ * 4, stream);
  hipMemsetAsync(counts, 0, (size_t)G_ * 4, stream);
  gp_count<<<(N_ + 255) / 256, 256, 0, stream>>>(batch, counts);
  gp_pool_sum<<<N_, 64, 0, stream>>>(h, batch, gemb);
  gp_pool_div<<<(G_ * HID_ + 255) / 256, 256, 0, stream>>>(gemb, counts);

  // ---- 2 GRU layers over the 64-long sequence ----
  gp_gru<<<1, HID_, 0, stream>>>(gemb, gru_wih, gru_whh, gru_bih, gru_bhh, seqB);
  gp_gru<<<1, HID_, 0, stream>>>(seqB,
      gru_wih + (size_t)3 * HID_ * HID_, gru_whh + (size_t)3 * HID_ * HID_,
      gru_bih + 3 * HID_, gru_bhh + 3 * HID_, seqC);

  // ---- classifier ----
  gp_classifier<<<G_, HID_ / 2, 0, stream>>>(seqC, cw1, cb1, cw2, cb2, out);

  (void)in_sizes; (void)n_in; (void)out_size; (void)ws_size;
}
